// Attention_85418309583085
// MI455X (gfx1250) — matrix operand order repaired and checked by reading, not compiled, not run on hardware
//
#include <hip/hip_runtime.h>
#include <hip/hip_bf16.h>
#include <math.h>

// ---------------------------------------------------------------------------
// Shapes (fixed): B=32, N=10, L=S=128, H=1024, BN=320, NL=1280
// d_out (flat, return order): out[32*1280*1024] | ex_attn[32*1280*1280] |
//                             set_attn[320*128*128]   (all f32)
//
// Strategy: all matmul FLOPs on v_wmma_f32_16x16x32_bf16 (f32 accum).
// One-time bf16 conversion (+ transposed copies) so every fragment load in a
// hot loop is two contiguous 16B bf16 runs with zero VALU converts. x is kept bf16 in
// workspace. Stage-2 x tiles staged into LDS via the Tensor Data Mover.
// ---------------------------------------------------------------------------

typedef __attribute__((ext_vector_type(16))) __bf16 v16bf;
typedef __attribute__((ext_vector_type(8)))  float  v8f;
typedef unsigned int u32x4 __attribute__((ext_vector_type(4)));
typedef int          i32x8 __attribute__((ext_vector_type(8)));
typedef int          i32x4 __attribute__((ext_vector_type(4)));

#define WMMA_BF16(A, Bm, C) \
  __builtin_amdgcn_wmma_f32_16x16x32_bf16(false, (A), false, (Bm), (short)0, (C), false, false)

// ---- fragment loaders (CDNA5 wave32 layouts, CDNA5 ISA §7.12.2) -------
// A (16x32): lane m=lane&15, kb=(lane>>4)*8;  elem e -> K = kb + e + (e>=8?8:0)
// B (32x16): lane n=lane&15, kb=(lane>>4)*8;  elem e -> K = kb + e + (e>=8?8:0)
// C/D (16x16 f32): elem r -> row r + 8*(lane>>4), col lane&15

// A fragment from bf16 (LDS or global; addrspace inferred) — 2x 16B loads
static __device__ inline v16bf frag_A_b(const __bf16* p0, int ld, int k0, int lane) {
  int m = lane & 15, kb = (lane >> 4) * 8;
  const __bf16* p = p0 + (size_t)m * ld + k0 + kb;
  v16bf a;
#pragma unroll
  for (int e = 0; e < 16; ++e) a[e] = p[e + ((e >= 8) ? 8 : 0)];
  return a;
}

// A fragment from f32 LDS (softmax probabilities) — converts on read
static __device__ inline v16bf frag_A_lf(const float* S, int ld, int k0, int lane) {
  int m = lane & 15, kb = (lane >> 4) * 8;
  const float* p = S + (size_t)m * ld + k0 + kb;
  v16bf a;
#pragma unroll
  for (int e = 0; e < 16; ++e) a[e] = (__bf16)p[e + ((e >= 8) ? 8 : 0)];
  return a;
}

// B fragment, B[k][n] = src[n0+n][k0+k] (bf16 row-major src) — 2x 16B loads
static __device__ inline v16bf frag_Bt_b(const __bf16* p0, int ld, int n0, int k0, int lane) {
  int n = lane & 15, kb = (lane >> 4) * 8;
  const __bf16* p = p0 + (size_t)(n0 + n) * ld + k0 + kb;
  v16bf b;
#pragma unroll
  for (int e = 0; e < 16; ++e) b[e] = p[e + ((e >= 8) ? 8 : 0)];
  return b;
}

// ---------------------------------------------------------------------------
// Pre-pass: f32 -> bf16 copy (n multiple of 8)
// ---------------------------------------------------------------------------
__global__ void cvt_bf16_kernel(const float* __restrict__ src, __bf16* __restrict__ dst,
                                long n) {
  long i = ((long)blockIdx.x * blockDim.x + threadIdx.x) * 8;
  if (i < n) {
#pragma unroll
    for (int e = 0; e < 8; ++e) dst[i + e] = (__bf16)src[i + e];
  }
}

// Pre-pass: f32 [batch][R][C] -> bf16 [batch][C][R] (LDS-tiled transpose)
__global__ void transpose_cvt_kernel(const float* __restrict__ src, __bf16* __restrict__ dst,
                                     int R, int C) {
  __shared__ float tile[32][33];
  int bb = blockIdx.z;
  int r0 = blockIdx.y * 32, c0 = blockIdx.x * 32;
  const float* s = src + (size_t)bb * R * C;
  __bf16* d = dst + (size_t)bb * R * C;
  int tx = threadIdx.x, ty = threadIdx.y;   // 32 x 8
#pragma unroll
  for (int i = ty; i < 32; i += 8)
    tile[i][tx] = s[(size_t)(r0 + i) * C + c0 + tx];
  __syncthreads();
#pragma unroll
  for (int i = ty; i < 32; i += 8)
    d[(size_t)(c0 + i) * R + r0 + tx] = (__bf16)tile[tx][i];
}

// ---------------------------------------------------------------------------
// Kernel 1: fused set-attention + tanh-GEMM.  One wave per (bn, 16-row tile).
// LDS: sP f32[16][128] (8KB) | sQ bf16[16][1024] (32KB) | sCtx bf16[16][1024] (32KB)
// ---------------------------------------------------------------------------
__global__ void stage1_kernel(const float* __restrict__ outp,
                              const __bf16* __restrict__ c1b,   // [32][128][1024]
                              const __bf16* __restrict__ c1t,   // [32][1024][128]
                              const __bf16* __restrict__ wsetb, // [1024][2048]
                              const float* __restrict__ bset,
                              float* __restrict__ set_attn,
                              __bf16* __restrict__ xb) {        // [320*128][1024]
  extern __shared__ char smem[];
  float*  sP   = (float*)smem;                      // 16*128 f32
  __bf16* sQ   = (__bf16*)(smem + 16 * 128 * 4);    // 16*1024
  __bf16* sCtx = sQ + 16 * 1024;                    // 16*1024

  const int lane = threadIdx.x & 31;
  const int wid  = blockIdx.x;        // 0..2559
  const int bn   = wid >> 3;
  const int l0   = (wid & 7) * 16;
  const int b    = bn / 10;

  const float*  Q   = outp + ((size_t)bn * 128 + l0) * 1024;
  const __bf16* K1  = c1b + (size_t)b * 128 * 1024;
  const __bf16* V1t = c1t + (size_t)b * 1024 * 128;

  for (int i = lane; i < 16 * 1024; i += 32) sQ[i] = (__bf16)Q[i];
  __syncthreads();

  v8f zero = {};
  // ---- scores = Q @ K1^T : 16x128 ----
  v8f acc[8];
#pragma unroll
  for (int st = 0; st < 8; ++st) acc[st] = zero;
  for (int kt = 0; kt < 32; ++kt) {
    v16bf a = frag_A_b(sQ, 1024, kt * 32, lane);
#pragma unroll
    for (int st = 0; st < 8; ++st) {
      v16bf bm = frag_Bt_b(K1, 1024, st * 16, kt * 32, lane);
      acc[st] = WMMA_BF16(a, bm, acc[st]);
    }
  }
  {
    int n = lane & 15, hi = lane >> 4;
#pragma unroll
    for (int st = 0; st < 8; ++st)
#pragma unroll
      for (int r = 0; r < 8; ++r)
        sP[(r + 8 * hi) * 128 + st * 16 + n] = acc[st][r];
  }
  __syncthreads();

  // ---- row softmax (lane i owns row i); emit set_attn ----
  if (lane < 16) {
    float* row = sP + lane * 128;
    float m = -3.4e38f;
    for (int i = 0; i < 128; ++i) m = fmaxf(m, row[i]);
    float s = 0.f;
    for (int i = 0; i < 128; ++i) { float e = __expf(row[i] - m); row[i] = e; s += e; }
    float inv = 1.f / s;
    float* orow = set_attn + ((size_t)bn * 128 + l0 + lane) * 128;
    for (int i = 0; i < 128; ++i) { float p = row[i] * inv; row[i] = p; orow[i] = p; }
  }
  __syncthreads();

  // ---- set_ctx = P @ V1 -> sCtx (bf16); V from transposed copy ----
  v16bf aP[4];
#pragma unroll
  for (int kt = 0; kt < 4; ++kt) aP[kt] = frag_A_lf(sP, 128, kt * 32, lane);
  for (int nt = 0; nt < 64; ++nt) {
    v8f c2 = zero;
#pragma unroll
    for (int kt = 0; kt < 4; ++kt) {
      v16bf bm = frag_Bt_b(V1t, 128, nt * 16, kt * 32, lane);
      c2 = WMMA_BF16(aP[kt], bm, c2);
    }
    int n = lane & 15, hi = lane >> 4;
#pragma unroll
    for (int r = 0; r < 8; ++r)
      sCtx[(r + 8 * hi) * 1024 + nt * 16 + n] = (__bf16)c2[r];
  }
  __syncthreads();

  // ---- x = tanh([Q | set_ctx] @ Wset^T + bset) -> x_bf (4 j-tiles / A-load) ----
  for (int jg = 0; jg < 16; ++jg) {
    v8f cx[4];
#pragma unroll
    for (int u = 0; u < 4; ++u) cx[u] = zero;
    for (int kt = 0; kt < 32; ++kt) {
      v16bf a = frag_A_b(sQ, 1024, kt * 32, lane);
#pragma unroll
      for (int u = 0; u < 4; ++u) {
        v16bf bm = frag_Bt_b(wsetb, 2048, (jg * 4 + u) * 16, kt * 32, lane);
        cx[u] = WMMA_BF16(a, bm, cx[u]);
      }
    }
    for (int kt = 0; kt < 32; ++kt) {
      v16bf a = frag_A_b(sCtx, 1024, kt * 32, lane);
#pragma unroll
      for (int u = 0; u < 4; ++u) {
        v16bf bm = frag_Bt_b(wsetb, 2048, (jg * 4 + u) * 16, 1024 + kt * 32, lane);
        cx[u] = WMMA_BF16(a, bm, cx[u]);
      }
    }
    int n = lane & 15, hi = lane >> 4;
#pragma unroll
    for (int u = 0; u < 4; ++u) {
      int j = (jg * 4 + u) * 16 + n;
      float bias = bset[j];
#pragma unroll
      for (int r = 0; r < 8; ++r)
        xb[((size_t)bn * 128 + l0 + r + 8 * hi) * 1024 + j] = (__bf16)tanhf(cx[u][r] + bias);
    }
  }
}

// ---------------------------------------------------------------------------
// Kernel 2: raw example scores = x @ ex_out^T. One wave per 16x64 tile strip.
// ---------------------------------------------------------------------------
__global__ void ex_scores_kernel(const __bf16* __restrict__ xb,
                                 const __bf16* __restrict__ c0b,
                                 float* __restrict__ exattn) {
  const int lane = threadIdx.x & 31;
  int wid = blockIdx.x;                 // 32*80*20
  int b   = wid / (80 * 20);
  int rem = wid % (80 * 20);
  int qt  = rem / 20, kg = rem % 20;

  const __bf16* X = xb  + ((size_t)b * 1280 + qt * 16) * 1024;
  const __bf16* K = c0b + ((size_t)b * 1280 + kg * 64) * 1024;

  v8f zero = {};
  v8f acc[4];
#pragma unroll
  for (int u = 0; u < 4; ++u) acc[u] = zero;
  for (int kk = 0; kk < 32; ++kk) {
    v16bf a = frag_A_b(X, 1024, kk * 32, lane);
#pragma unroll
    for (int u = 0; u < 4; ++u) {
      v16bf bm = frag_Bt_b(K, 1024, u * 16, kk * 32, lane);
      acc[u] = WMMA_BF16(a, bm, acc[u]);
    }
  }
  int n = lane & 15, hi = lane >> 4;
#pragma unroll
  for (int u = 0; u < 4; ++u)
#pragma unroll
    for (int r = 0; r < 8; ++r)
      exattn[((size_t)b * 1280 + qt * 16 + r + 8 * hi) * 1280 + kg * 64 + u * 16 + n] = acc[u][r];
}

// ---------------------------------------------------------------------------
// Kernel 3: in-place row softmax over 1280 columns (40960 rows).
// ---------------------------------------------------------------------------
__global__ void row_softmax_kernel(float* __restrict__ buf) {
  __shared__ float red[256];
  float* row = buf + (size_t)blockIdx.x * 1280;
  const int t = threadIdx.x;

  float m = -3.4e38f;
  for (int i = t; i < 1280; i += 256) m = fmaxf(m, row[i]);
  red[t] = m;
  __syncthreads();
  for (int s = 128; s > 0; s >>= 1) {
    if (t < s) red[t] = fmaxf(red[t], red[t + s]);
    __syncthreads();
  }
  m = red[0];
  __syncthreads();

  float s = 0.f;
  for (int i = t; i < 1280; i += 256) s += __expf(row[i] - m);
  red[t] = s;
  __syncthreads();
  for (int s2 = 128; s2 > 0; s2 >>= 1) {
    if (t < s2) red[t] += red[t + s2];
    __syncthreads();
  }
  float inv = 1.f / red[0];

  for (int i = t; i < 1280; i += 256) row[i] = __expf(row[i] - m) * inv;
}

// ---------------------------------------------------------------------------
// Kernel 4: fused ex_ctx = ex_attn @ ex_out, out = tanh([x|ex_ctx]@Wout^T + b).
// One wave per (b, 16-row q tile).  x tile staged into LDS by the TDM.
// LDS: sP bf16[16][1280] (40KB, off 0) | sX bf16[16][1024] (32KB, off 40960)
//      | sCtx bf16[16][1024] (32KB, off 73728)
// ---------------------------------------------------------------------------
__global__ void stage2_kernel(const __bf16* __restrict__ xb,
                              const __bf16* __restrict__ c0t,   // [32][1024][1280]
                              const float* __restrict__ exattn,
                              const __bf16* __restrict__ woutb, // [1024][2048]
                              const float* __restrict__ bout,
                              float* __restrict__ outp) {
  extern __shared__ char smem[];
  __bf16* sP   = (__bf16*)smem;          // 16*1280
  __bf16* sX   = sP + 16 * 1280;         // 16*1024  (LDS byte offset 40960)
  __bf16* sCtx = sX + 16 * 1024;         // 16*1024

  const int lane = threadIdx.x & 31;
  int wid = blockIdx.x;                  // 32*80
  int b = wid / 80, q0 = (wid % 80) * 16;

  const float*  Prow = exattn + ((size_t)b * 1280 + q0) * 1280;
  const __bf16* Xrow = xb + ((size_t)b * 1280 + q0) * 1024;
  const __bf16* Vt   = c0t + (size_t)b * 1024 * 1280;

#if __has_builtin(__builtin_amdgcn_tensor_load_to_lds) && __has_builtin(__builtin_amdgcn_s_wait_tensorcnt)
  // --- TDM: DMA the contiguous 16x1024 bf16 x-tile into LDS (1-D tile) ---
  // D# group0/1 packed per CDNA5 ISA.
  {
    unsigned long long ga = (unsigned long long)Xrow;
    u32x4 g0;
    g0[0] = 1u;                                        // count=1, user mode
    g0[1] = 40960u;                                    // lds_addr (bytes) = sX
    g0[2] = (unsigned)ga;                              // global_addr[31:0]
    g0[3] = (unsigned)(ga >> 32) | (2u << 30);         // global_addr[56:32] | type=2
    i32x8 g1;
    g1[0] = 1 << 16;                                   // data_size=1 (2 bytes)
    g1[1] = 0x4000 << 16;                              // tensor_dim0 = 16384 (lo16)
    g1[2] = 1 << 16;                                   // tensor_dim0 hi=0 | tensor_dim1=1 (lo16)
    g1[3] = 0x4000 << 16;                              // tensor_dim1 hi=0 | tile_dim0 = 16384
    g1[4] = 0;                                         // tile_dim1=0, tile_dim2=0
    g1[5] = 16384;                                     // tensor_dim0_stride (lo32)
    g1[6] = 0;
    g1[7] = 0;
    i32x4 z4 = {0, 0, 0, 0};
    i32x8 z8 = {0, 0, 0, 0, 0, 0, 0, 0};
    __builtin_amdgcn_tensor_load_to_lds(g0, g1, z4, z4, z8, 0);
  }
#else
  for (int i = lane; i < 16 * 1024; i += 32) sX[i] = Xrow[i];
#endif

  // attention-probability tile -> LDS (f32 -> bf16, converted once)
  for (int i = lane; i < 16 * 1280; i += 32) sP[i] = (__bf16)Prow[i];
  __syncthreads();
#if __has_builtin(__builtin_amdgcn_tensor_load_to_lds) && __has_builtin(__builtin_amdgcn_s_wait_tensorcnt)
  __builtin_amdgcn_s_wait_tensorcnt(0);
#endif

  v8f zero = {};
  // ---- ex_ctx = P @ ex_out -> sCtx (4 n-tiles per A-load) ----
  for (int ng = 0; ng < 16; ++ng) {
    v8f c[4];
#pragma unroll
    for (int u = 0; u < 4; ++u) c[u] = zero;
    for (int kt = 0; kt < 40; ++kt) {
      v16bf a = frag_A_b(sP, 1280, kt * 32, lane);
#pragma unroll
      for (int u = 0; u < 4; ++u) {
        v16bf bm = frag_Bt_b(Vt, 1280, (ng * 4 + u) * 16, kt * 32, lane);
        c[u] = WMMA_BF16(a, bm, c[u]);
      }
    }
    int n = lane & 15, hi = lane >> 4;
#pragma unroll
    for (int u = 0; u < 4; ++u)
#pragma unroll
      for (int r = 0; r < 8; ++r)
        sCtx[(r + 8 * hi) * 1024 + (ng * 4 + u) * 16 + n] = (__bf16)c[u][r];
  }
  __syncthreads();

  // ---- out = tanh([x | ex_ctx] @ Wout^T + bout) (4 j-tiles per A-load) ----
  for (int jg = 0; jg < 16; ++jg) {
    if (jg + 1 < 16)
      __builtin_prefetch(woutb + (size_t)(jg + 1) * 4 * 16 * 2048, 0, 1);
    v8f c[4];
#pragma unroll
    for (int u = 0; u < 4; ++u) c[u] = zero;
    for (int kt = 0; kt < 32; ++kt) {
      v16bf a = frag_A_b(sX, 1024, kt * 32, lane);
#pragma unroll
      for (int u = 0; u < 4; ++u) {
        v16bf bm = frag_Bt_b(woutb, 2048, (jg * 4 + u) * 16, kt * 32, lane);
        c[u] = WMMA_BF16(a, bm, c[u]);
      }
    }
    for (int kt = 0; kt < 32; ++kt) {
      v16bf a = frag_A_b(sCtx, 1024, kt * 32, lane);
#pragma unroll
      for (int u = 0; u < 4; ++u) {
        v16bf bm = frag_Bt_b(woutb, 2048, (jg * 4 + u) * 16, 1024 + kt * 32, lane);
        c[u] = WMMA_BF16(a, bm, c[u]);
      }
    }
    int n = lane & 15, hi = lane >> 4;
#pragma unroll
    for (int u = 0; u < 4; ++u) {
      int j = (jg * 4 + u) * 16 + n;
      float bias = bout[j];
#pragma unroll
      for (int r = 0; r < 8; ++r)
        outp[((size_t)b * 1280 + q0 + r + 8 * hi) * 1024 + j] = tanhf(c[u][r] + bias);
    }
  }
}

// ---------------------------------------------------------------------------
extern "C" void kernel_launch(void* const* d_in, const int* in_sizes, int n_in,
                              void* d_out, int out_size, void* d_ws, size_t ws_size,
                              hipStream_t stream) {
  const float* outp = (const float*)d_in[0];   // output   [320,128,1024]
  const float* ctx0 = (const float*)d_in[1];   // context0 [320,128,1024]
  const float* ctx1 = (const float*)d_in[2];   // context1 [32,128,1024]
  // d_in[3] = mask (no effect on math per reference)
  const float* Wset = (const float*)d_in[4];   // [1024,2048]
  const float* bset = (const float*)d_in[5];   // [1024]
  const float* Wout = (const float*)d_in[6];   // [1024,2048]
  const float* bout = (const float*)d_in[7];   // [1024]

  float* out_final = (float*)d_out;                          // 32*1280*1024
  float* exattn    = out_final + (size_t)32 * 1280 * 1024;   // 32*1280*1280
  float* setattn   = exattn    + (size_t)32 * 1280 * 1280;   // 320*128*128

  // Workspace partition (bf16)
  __bf16* ws      = (__bf16*)d_ws;
  __bf16* x_bf    = ws;                                      // 41,943,040
  __bf16* c0_bf   = x_bf   + 41943040LL;                     // 41,943,040
  __bf16* c0t_bf  = c0_bf  + 41943040LL;                     // 41,943,040
  __bf16* c1_bf   = c0t_bf + 41943040LL;                     //  4,194,304
  __bf16* c1t_bf  = c1_bf  + 4194304LL;                      //  4,194,304
  __bf16* wset_bf = c1t_bf + 4194304LL;                      //  2,097,152
  __bf16* wout_bf = wset_bf + 2097152LL;                     //  2,097,152

  // ---- one-time bf16 conversion / transposition passes ----
  cvt_bf16_kernel<<<20480, 256, 0, stream>>>(ctx0, c0_bf, 41943040L);
  cvt_bf16_kernel<<<2048, 256, 0, stream>>>(ctx1, c1_bf, 4194304L);
  cvt_bf16_kernel<<<1024, 256, 0, stream>>>(Wset, wset_bf, 2097152L);
  cvt_bf16_kernel<<<1024, 256, 0, stream>>>(Wout, wout_bf, 2097152L);
  transpose_cvt_kernel<<<dim3(32, 40, 32), dim3(32, 8), 0, stream>>>(ctx0, c0t_bf, 1280, 1024);
  transpose_cvt_kernel<<<dim3(32, 4, 32), dim3(32, 8), 0, stream>>>(ctx1, c1t_bf, 128, 1024);

  // ---- stage 1: set attention + tanh GEMM (writes set_attn, x_bf) ----
  stage1_kernel<<<2560, 32, 73728, stream>>>(outp, c1_bf, c1t_bf, wset_bf, bset,
                                             setattn, x_bf);
  // ---- stage 2a: raw example scores into ex_attn region ----
  ex_scores_kernel<<<51200, 32, 0, stream>>>(x_bf, c0_bf, exattn);
  // ---- stage 2b: in-place softmax -> final ex_attn ----
  row_softmax_kernel<<<40960, 256, 0, stream>>>(exattn);
  // ---- stage 2c: ex_ctx + output tanh GEMM (writes out) ----
  stage2_kernel<<<2560, 32, 106496, stream>>>(x_bf, c0t_bf, exattn, wout_bf, bout, out_final);
}
